// TransformerBackbone_81990925681358
// MI455X (gfx1250) — matrix operand order repaired and checked by reading, not compiled, not run on hardware
//
#include <hip/hip_runtime.h>
#include <hip/hip_bf16.h>

// ---------------------------------------------------------------------------
// Types & helpers
// ---------------------------------------------------------------------------
typedef __attribute__((ext_vector_type(16))) __bf16         v16bf;
typedef __attribute__((ext_vector_type(8)))  float          v8f;
typedef __attribute__((ext_vector_type(8)))  unsigned short u16x8;
typedef __attribute__((ext_vector_type(4)))  unsigned int   v4u;
typedef __attribute__((ext_vector_type(8)))  int            v8i;
typedef __attribute__((ext_vector_type(4)))  int            v4i;

union BFrag { v16bf v; u16x8 h2[2]; };

__device__ __forceinline__ unsigned short f2bf(float f) {
    unsigned int u = __float_as_uint(f);
    unsigned int r = (u + 0x7FFFu + ((u >> 16) & 1u)) >> 16;  // RNE
    return (unsigned short)r;
}
__device__ __forceinline__ float bf2f(unsigned short h) {
    return __uint_as_float(((unsigned int)h) << 16);
}

#define NB   512      // batch (graphs)
#define SS   64       // max atoms (sequence)
#define HH   256      // hidden
#define NHD  8        // heads
#define HDH  32       // head dim
#define NL   6        // layers
#define MM   (NB*SS)  // padded rows = 32768

#define OUT_F32  0
#define OUT_BF16 1
#define OUT_GELU 2

// Tensor Data Mover availability (device pass only; 6-arg clang-23 form)
#if defined(__HIP_DEVICE_COMPILE__) && __has_builtin(__builtin_amdgcn_tensor_load_to_lds)
#define USE_TDM 1
#else
#define USE_TDM 0
#endif

#if defined(__HIP_DEVICE_COMPILE__)
__device__ __forceinline__ void wait_tensorcnt0() {
#if __has_builtin(__builtin_amdgcn_s_wait_tensorcnt)
    __builtin_amdgcn_s_wait_tensorcnt(0);
#else
    asm volatile("s_wait_tensorcnt 0x0" ::: "memory");
#endif
}
#else
__device__ __forceinline__ void wait_tensorcnt0() {}
#endif

#if USE_TDM
// 2D tile DMA: rows x 32 bf16 elements, global row stride = strideElems,
// destination LDS rows padded to pitch 40 ushort (32 data + 8 pad =
// pad_interval 16 DWORDs, pad_amount 4 DWORDs).
__device__ __forceinline__ void tdm_load_tile(unsigned lds_off,
                                              const unsigned short* gptr,
                                              unsigned rows,
                                              unsigned strideElems) {
    unsigned long long ga = (unsigned long long)(size_t)gptr;
    v4u g0;
    g0.x = 1u;                                    // count=1, user descriptor
    g0.y = lds_off;                               // lds_addr
    g0.z = (unsigned)(ga & 0xFFFFFFFFu);          // global_addr[31:0]
    g0.w = (unsigned)((ga >> 32) & 0x01FFFFFFu)   // global_addr[56:32]
         | 0x80000000u;                           // type=2 ("image")
    v8i g1;
    g1[0] = (int)((1u << 16)       // data_size = 2 bytes
                | (1u << 20)       // pad_enable
                | (3u << 22)       // pad_interval: 16 DWORDs
                | (3u << 25));     // pad_amount: 4 DWORDs
    g1[1] = (int)(32u << 16);                       // tensor_dim0[15:0] = 32
    g1[2] = (int)((rows & 0xFFFFu) << 16);          // tensor_dim1[15:0]
    g1[3] = (int)(32u << 16);                       // tile_dim0 = 32
    g1[4] = (int)(rows & 0xFFFFu);                  // tile_dim1 = rows
    g1[5] = (int)strideElems;                       // tensor_dim0_stride lo32
    g1[6] = 0;
    g1[7] = 0;
    v4i gz4 = {0, 0, 0, 0};
    v8i gz8 = {0, 0, 0, 0, 0, 0, 0, 0};
    __builtin_amdgcn_tensor_load_to_lds(g0, g1, gz4, gz4, gz8, 0);
}
#endif

// ---------------------------------------------------------------------------
// f32 -> bf16 weight conversion
// ---------------------------------------------------------------------------
__global__ void cvt_bf16_kernel(const float* __restrict__ src,
                                unsigned short* __restrict__ dst, int n) {
    int i = blockIdx.x * 256 + threadIdx.x;
    if (i < n) dst[i] = f2bf(src[i]);
}

__global__ void zero_kernel(float* __restrict__ xp,
                            unsigned short* __restrict__ xbf, long long n) {
    long long i = (long long)blockIdx.x * 256 + threadIdx.x;
    if (i < n) { xp[i] = 0.0f; xbf[i] = 0; }
}

// exclusive prefix sum of counts[512] -> offs[512]
__global__ void scan_counts_kernel(const int* __restrict__ counts,
                                   int* __restrict__ offs) {
    __shared__ int s[NB];
    int t = threadIdx.x;
    s[t] = counts[t];
    __syncthreads();
    for (int d = 1; d < NB; d <<= 1) {
        int add = (t >= d) ? s[t - d] : 0;
        __syncthreads();
        s[t] += add;
        __syncthreads();
    }
    offs[t] = s[t] - counts[t];
}

// ---------------------------------------------------------------------------
// Time embedding: sinusoidal -> MLP(128->128, silu) -> 128 -> proj 128->256
// ---------------------------------------------------------------------------
__global__ __launch_bounds__(128)
void time_kernel(const float* __restrict__ t, const float* __restrict__ inv_freq,
                 const float* __restrict__ te_W1, const float* __restrict__ te_b1,
                 const float* __restrict__ te_W2, const float* __restrict__ te_b2,
                 const float* __restrict__ tp_W,  const float* __restrict__ tp_b,
                 float* __restrict__ t_emb) {
    __shared__ float emb[128], e1[128], e2[128];
    int b = blockIdx.x, j = threadIdx.x;
    if (j < 64) {
        float s = t[b] * inv_freq[j];
        emb[j]      = sinf(s);
        emb[j + 64] = cosf(s);
    }
    __syncthreads();
    float a = te_b1[j];
    for (int k = 0; k < 128; ++k) a += emb[k] * te_W1[j * 128 + k];
    e1[j] = a / (1.0f + expf(-a));  // silu
    __syncthreads();
    float c = te_b2[j];
    for (int k = 0; k < 128; ++k) c += e1[k] * te_W2[j * 128 + k];
    e2[j] = c;
    __syncthreads();
    for (int h = j; h < HH; h += 128) {
        float o = tp_b[h];
        for (int k = 0; k < 128; ++k) o += e2[k] * tp_W[h * 128 + k];
        t_emb[(size_t)b * HH + h] = o;
    }
}

// ---------------------------------------------------------------------------
// Node embedding: softmax(theta) @ node_W^T + pos-MLP + t_emb, scatter padded
// ---------------------------------------------------------------------------
__global__ __launch_bounds__(256)
void embed_kernel(const float* __restrict__ theta_t, const float* __restrict__ pos_t,
                  const int* __restrict__ batch, const int* __restrict__ offs,
                  const float* __restrict__ node_W, const float* __restrict__ node_b,
                  const float* __restrict__ pe_W1, const float* __restrict__ pe_b1,
                  const float* __restrict__ pe_W2, const float* __restrict__ pe_b2,
                  const float* __restrict__ t_emb,
                  float* __restrict__ xp, unsigned short* __restrict__ xbf) {
    __shared__ float sm[16];
    __shared__ float pd[3];
    __shared__ float p1[256];
    int i = blockIdx.x, j = threadIdx.x;
    if (j < 16) sm[j] = theta_t[(size_t)i * 16 + j];
    if (j < 3)  pd[j] = pos_t[(size_t)i * 3 + j];
    __syncthreads();
    float mx = sm[0];
    for (int a = 1; a < 16; ++a) mx = fmaxf(mx, sm[a]);
    float e[16]; float se = 0.0f;
    for (int a = 0; a < 16; ++a) { e[a] = expf(sm[a] - mx); se += e[a]; }
    float inv = 1.0f / se;
    float x = node_b[j];
    for (int a = 0; a < 16; ++a) x += e[a] * inv * node_W[j * 16 + a];
    float h1 = pe_b1[j];
    for (int d = 0; d < 3; ++d) h1 += pd[d] * pe_W1[j * 3 + d];
    p1[j] = h1 / (1.0f + expf(-h1));
    __syncthreads();
    float h2 = pe_b2[j];
    for (int k = 0; k < 256; ++k) h2 += p1[k] * pe_W2[j * 256 + k];
    int b   = batch[i];
    int pos = i - offs[b];
    x += h2 + t_emb[(size_t)b * HH + j];
    size_t row = (size_t)b * SS + pos;
    xp[row * HH + j]  = x;
    xbf[row * HH + j] = f2bf(x);
}

// ---------------------------------------------------------------------------
// Generic bf16 WMMA GEMM: C[M,N] = A[M,K] * W[N,K]^T + bias[N]
// Block tile 128x64; 8 waves (4m x 2n), each wave 32x32 = 4 accumulators.
// Double-buffered LDS (pitch 40 ushort); staged via TDM when available,
// else VGPR-prefetch software pipeline. One barrier per k-step.
// ---------------------------------------------------------------------------
__global__ __launch_bounds__(256)
void gemm_bf16_kernel(const unsigned short* __restrict__ A,
                      const unsigned short* __restrict__ W,
                      const float* __restrict__ bias,
                      void* __restrict__ out,
                      int N, int K, int mode) {
    __shared__ __align__(16) unsigned short Alds[2][128 * 40];
    __shared__ __align__(16) unsigned short Blds[2][64 * 40];
    const int t    = threadIdx.x;
    const int lane = t & 31;
    const int w    = t >> 5;        // 0..7
    const int l16  = lane & 15;
    const int hh   = lane >> 4;     // wave half
    const int mblk = blockIdx.x * 128;
    const int nblk = blockIdx.y * 64;

    // cooperative-load addressing (fallback path)
    const int arow = t >> 1;            // 0..127
    const int acol = (t & 1) * 16;      // 0,16
    const int brow = t >> 2;            // 0..63
    const int bcol = (t & 3) * 8;       // 0..24
    const unsigned short* gA = &A[(size_t)(mblk + arow) * K + acol];
    const unsigned short* gW = &W[(size_t)(nblk + brow) * K + bcol];

    // ---- stage k0 = 0 into buffer 0 ----
#if USE_TDM
    if (w == 0) {
        tdm_load_tile((unsigned)(size_t)&Alds[0][0], A + (size_t)mblk * K, 128, K);
        tdm_load_tile((unsigned)(size_t)&Blds[0][0], W + (size_t)nblk * K, 64, K);
        wait_tensorcnt0();
    }
#else
    {
        u16x8 a0 = *(const u16x8*)(gA);
        u16x8 a1 = *(const u16x8*)(gA + 8);
        u16x8 b0 = *(const u16x8*)(gW);
        *(u16x8*)&Alds[0][arow * 40 + acol]     = a0;
        *(u16x8*)&Alds[0][arow * 40 + acol + 8] = a1;
        *(u16x8*)&Blds[0][brow * 40 + bcol]     = b0;
    }
#endif
    __syncthreads();

    v8f acc00 = {0,0,0,0,0,0,0,0};
    v8f acc01 = {0,0,0,0,0,0,0,0};
    v8f acc10 = {0,0,0,0,0,0,0,0};
    v8f acc11 = {0,0,0,0,0,0,0,0};

    const int mw  = 32 * (w & 3);
    const int nw  = 32 * (w >> 2);
    const int klo = hh * 16;
    const int kb  = hh * 16;
    int cur = 0;

    for (int k0 = 0; k0 < K; k0 += 32) {
        const bool last = (k0 + 32 >= K);
        const int  nxt  = cur ^ 1;

        // kick off next tile while we compute this one
#if USE_TDM
        if (!last && w == 0) {
            tdm_load_tile((unsigned)(size_t)&Alds[nxt][0],
                          A + (size_t)mblk * K + (k0 + 32), 128, K);
            tdm_load_tile((unsigned)(size_t)&Blds[nxt][0],
                          W + (size_t)nblk * K + (k0 + 32), 64, K);
        }
#else
        u16x8 pa0, pa1, pb;
        if (!last) {
            const unsigned short* nA = gA + (k0 + 32);
            const unsigned short* nW = gW + (k0 + 32);
            pa0 = *(const u16x8*)(nA);
            pa1 = *(const u16x8*)(nA + 8);
            pb  = *(const u16x8*)(nW);
        }
#endif
        // compute from current buffer
        const unsigned short* Ab = &Alds[cur][0];
        const unsigned short* Bb = &Blds[cur][0];
        BFrag a0, a1, b0, b1;
        a0.h2[0] = *(const u16x8*)&Ab[(mw + l16) * 40 + klo];
        a0.h2[1] = *(const u16x8*)&Ab[(mw + l16) * 40 + 8 + klo];
        a1.h2[0] = *(const u16x8*)&Ab[(mw + 16 + l16) * 40 + klo];
        a1.h2[1] = *(const u16x8*)&Ab[(mw + 16 + l16) * 40 + 8 + klo];
        b0.h2[0] = *(const u16x8*)&Bb[(nw + l16) * 40 + kb];
        b0.h2[1] = *(const u16x8*)&Bb[(nw + l16) * 40 + kb + 8];
        b1.h2[0] = *(const u16x8*)&Bb[(nw + 16 + l16) * 40 + kb];
        b1.h2[1] = *(const u16x8*)&Bb[(nw + 16 + l16) * 40 + kb + 8];

        acc00 = __builtin_amdgcn_wmma_f32_16x16x32_bf16(
            false, a0.v, false, b0.v, (short)0, acc00, false, false);
        acc01 = __builtin_amdgcn_wmma_f32_16x16x32_bf16(
            false, a0.v, false, b1.v, (short)0, acc01, false, false);
        acc10 = __builtin_amdgcn_wmma_f32_16x16x32_bf16(
            false, a1.v, false, b0.v, (short)0, acc10, false, false);
        acc11 = __builtin_amdgcn_wmma_f32_16x16x32_bf16(
            false, a1.v, false, b1.v, (short)0, acc11, false, false);

        if (!last) {
#if USE_TDM
            if (w == 0) wait_tensorcnt0();
#else
            *(u16x8*)&Alds[nxt][arow * 40 + acol]     = pa0;
            *(u16x8*)&Alds[nxt][arow * 40 + acol + 8] = pa1;
            *(u16x8*)&Blds[nxt][brow * 40 + bcol]     = pb;
#endif
            __syncthreads();
            cur = nxt;
        }
    }

    // epilogue
    const int mb  = mblk + 32 * (w & 3) + hh * 8;
    const int nb  = nblk + 32 * (w >> 2) + l16;
    auto emit = [&](v8f acc, int m0, int col) {
        float bv = bias ? bias[col] : 0.0f;
        for (int r = 0; r < 8; ++r) {
            float val = acc[r] + bv;
            size_t idx = (size_t)(m0 + r) * N + col;
            if (mode == OUT_F32) {
                ((float*)out)[idx] = val;
            } else if (mode == OUT_BF16) {
                ((unsigned short*)out)[idx] = f2bf(val);
            } else {
                float g = 0.5f * val * (1.0f + erff(val * 0.70710678118654752f));
                ((unsigned short*)out)[idx] = f2bf(g);
            }
        }
    };
    emit(acc00, mb,      nb);
    emit(acc01, mb,      nb + 16);
    emit(acc10, mb + 16, nb);
    emit(acc11, mb + 16, nb + 16);
}

// ---------------------------------------------------------------------------
// Attention: one block per (head, graph), 128 threads (4 waves).
// ---------------------------------------------------------------------------
__global__ __launch_bounds__(128)
void attn_kernel(const unsigned short* __restrict__ qkv,
                 const int* __restrict__ counts,
                 unsigned short* __restrict__ obf) {
    __shared__ float Sld[64 * 65];                        // scores (pitch 65)
    __shared__ __align__(16) unsigned short Pld[64 * 72]; // probs bf16
    __shared__ __align__(16) unsigned short Vt[32 * 72];  // V transposed
    const int h = blockIdx.x;
    const int b = blockIdx.y;
    const int t = threadIdx.x;
    const int lane = t & 31;
    const int w = t >> 5;        // 0..3 -> query rows 16w..
    const int l16 = lane & 15;
    const int hh = lane >> 4;

    const unsigned short* qbase = qkv + (size_t)b * SS * 768 + h * HDH;

    BFrag a;
    {
        int m = 16 * w + l16;
        int klo = hh * 16;
        const unsigned short* qr = qbase + (size_t)m * 768;
        a.h2[0] = *(const u16x8*)(qr + klo);
        a.h2[1] = *(const u16x8*)(qr + 8 + klo);
    }
    const float scale = 0.17677669529663687f;  // 1/sqrt(32)
    for (int c = 0; c < 4; ++c) {
        BFrag bf;
        int n = c * 16 + l16;
        const unsigned short* kr = qbase + 256 + (size_t)n * 768;
        int kb = hh * 16;
        bf.h2[0] = *(const u16x8*)(kr + kb);
        bf.h2[1] = *(const u16x8*)(kr + kb + 8);
        v8f acc = {0,0,0,0,0,0,0,0};
        acc = __builtin_amdgcn_wmma_f32_16x16x32_bf16(
            false, a.v, false, bf.v, (short)0, acc, false, false);
        int mloc = 16 * w + hh * 8;
        for (int r = 0; r < 8; ++r)
            Sld[(mloc + r) * 65 + c * 16 + l16] = acc[r] * scale;
    }
    {
        int kr = t >> 1;
        int d0 = (t & 1) * 16;
        const unsigned short* vr = qbase + 512 + (size_t)kr * 768 + d0;
        u16x8 v0 = *(const u16x8*)(vr);
        u16x8 v1 = *(const u16x8*)(vr + 8);
        for (int e2 = 0; e2 < 8; ++e2) {
            Vt[(d0 + e2) * 72 + kr]     = v0[e2];
            Vt[(d0 + 8 + e2) * 72 + kr] = v1[e2];
        }
    }
    __syncthreads();
    if (t < 64) {
        int cnt = counts[b];
        float mx = -1e30f;
        for (int c = 0; c < cnt; ++c) mx = fmaxf(mx, Sld[t * 65 + c]);
        float sum = 0.0f;
        for (int c = 0; c < 64; ++c) {
            float ev = (c < cnt) ? expf(Sld[t * 65 + c] - mx) : 0.0f;
            Sld[t * 65 + c] = ev;
            sum += ev;
        }
        float inv = 1.0f / sum;
        for (int c = 0; c < 64; ++c)
            Pld[t * 72 + c] = f2bf(Sld[t * 65 + c] * inv);
    }
    __syncthreads();
    v8f o0 = {0,0,0,0,0,0,0,0};
    v8f o1 = {0,0,0,0,0,0,0,0};
    for (int k0 = 0; k0 < 64; k0 += 32) {
        BFrag ap;
        {
            int m = 16 * w + l16;
            int klo = hh * 16;
            ap.h2[0] = *(const u16x8*)&Pld[m * 72 + k0 + klo];
            ap.h2[1] = *(const u16x8*)&Pld[m * 72 + k0 + 8 + klo];
        }
        int kb = hh * 16;
        BFrag b0, b1;
        b0.h2[0] = *(const u16x8*)&Vt[l16 * 72 + k0 + kb];
        b0.h2[1] = *(const u16x8*)&Vt[l16 * 72 + k0 + kb + 8];
        b1.h2[0] = *(const u16x8*)&Vt[(16 + l16) * 72 + k0 + kb];
        b1.h2[1] = *(const u16x8*)&Vt[(16 + l16) * 72 + k0 + kb + 8];
        o0 = __builtin_amdgcn_wmma_f32_16x16x32_bf16(
            false, ap.v, false, b0.v, (short)0, o0, false, false);
        o1 = __builtin_amdgcn_wmma_f32_16x16x32_bf16(
            false, ap.v, false, b1.v, (short)0, o1, false, false);
    }
    int mBase = 16 * w + hh * 8;
    for (int j = 0; j < 2; ++j) {
        v8f acc = j ? o1 : o0;
        int col = h * HDH + j * 16 + l16;
        for (int r = 0; r < 8; ++r) {
            size_t row = (size_t)b * SS + mBase + r;
            obf[row * HH + col] = f2bf(acc[r]);
        }
    }
}

// ---------------------------------------------------------------------------
// Residual + LayerNorm over H=256: xp = LN(xp + y); also emit bf16 mirror
// ---------------------------------------------------------------------------
__global__ __launch_bounds__(256)
void resid_ln_kernel(float* __restrict__ xp, const float* __restrict__ y,
                     const float* __restrict__ g, const float* __restrict__ bb,
                     unsigned short* __restrict__ xbf) {
    __shared__ float red[256];
    int r = blockIdx.x, j = threadIdx.x;
    size_t idx = (size_t)r * HH + j;
    float v = xp[idx] + y[idx];
    red[j] = v;
    __syncthreads();
    for (int s = 128; s > 0; s >>= 1) {
        if (j < s) red[j] += red[j + s];
        __syncthreads();
    }
    float mu = red[0] * (1.0f / 256.0f);
    __syncthreads();
    float d = v - mu;
    red[j] = d * d;
    __syncthreads();
    for (int s = 128; s > 0; s >>= 1) {
        if (j < s) red[j] += red[j + s];
        __syncthreads();
    }
    float var = red[0] * (1.0f / 256.0f);
    float o = d * rsqrtf(var + 1e-5f) * g[j] + bb[j];
    xp[idx]  = o;
    xbf[idx] = f2bf(o);
}

// ---------------------------------------------------------------------------
// Masked mean pool + head: out = silu(pooled @ pool_W^T + pool_b)
// ---------------------------------------------------------------------------
__global__ __launch_bounds__(256)
void pool_kernel(const float* __restrict__ xp, const int* __restrict__ counts,
                 const float* __restrict__ pool_W, const float* __restrict__ pool_b,
                 float* __restrict__ out) {
    __shared__ float pooled[256];
    int b = blockIdx.x, j = threadIdx.x;
    int cnt = counts[b];
    float s = 0.0f;
    for (int a = 0; a < cnt; ++a) s += xp[((size_t)b * SS + a) * HH + j];
    pooled[j] = s / (float)cnt;
    __syncthreads();
    float o = pool_b[j];
    for (int k = 0; k < 256; ++k) o += pooled[k] * pool_W[j * 256 + k];
    out[(size_t)b * HH + j] = o / (1.0f + expf(-o));
}

// ---------------------------------------------------------------------------
// Host orchestration
// ---------------------------------------------------------------------------
extern "C" void kernel_launch(void* const* d_in, const int* in_sizes, int n_in,
                              void* d_out, int out_size, void* d_ws, size_t ws_size,
                              hipStream_t stream) {
    (void)n_in; (void)out_size; (void)ws_size;
    const float* theta_t = (const float*)d_in[0];
    const float* pos_t   = (const float*)d_in[1];
    const float* t_in    = (const float*)d_in[2];
    const int*   batch   = (const int*)d_in[3];
    const int*   counts  = (const int*)d_in[4];
    const float* node_W  = (const float*)d_in[5];
    const float* node_b  = (const float*)d_in[6];
    const float* pe_W1   = (const float*)d_in[7];
    const float* pe_b1   = (const float*)d_in[8];
    const float* pe_W2   = (const float*)d_in[9];
    const float* pe_b2   = (const float*)d_in[10];
    const float* inv_freq= (const float*)d_in[11];
    const float* te_W1   = (const float*)d_in[12];
    const float* te_b1   = (const float*)d_in[13];
    const float* te_W2   = (const float*)d_in[14];
    const float* te_b2   = (const float*)d_in[15];
    const float* tp_W    = (const float*)d_in[16];
    const float* tp_b    = (const float*)d_in[17];
    const float* Wqkv    = (const float*)d_in[18];
    const float* bqkv    = (const float*)d_in[19];
    const float* Wo      = (const float*)d_in[20];
    const float* bo      = (const float*)d_in[21];
    const float* ln1_g   = (const float*)d_in[22];
    const float* ln1_b   = (const float*)d_in[23];
    const float* f_W1    = (const float*)d_in[24];
    const float* f_b1    = (const float*)d_in[25];
    const float* f_W2    = (const float*)d_in[26];
    const float* f_b2    = (const float*)d_in[27];
    const float* ln2_g   = (const float*)d_in[28];
    const float* ln2_b   = (const float*)d_in[29];
    const float* pool_W  = (const float*)d_in[30];
    const float* pool_b  = (const float*)d_in[31];

    const int N = in_sizes[0] / 16;  // number of atoms

    char* p = (char*)d_ws;
    auto alloc = [&](size_t bytes) -> char* {
        char* r = p;
        p += (bytes + 255) & ~(size_t)255;
        return r;
    };
    int*            offs    = (int*)alloc(NB * sizeof(int));
    float*          t_emb   = (float*)alloc((size_t)NB * HH * sizeof(float));
    unsigned short* wqkv_bf = (unsigned short*)alloc((size_t)NL * 768 * 256 * 2);
    unsigned short* wo_bf   = (unsigned short*)alloc((size_t)NL * 256 * 256 * 2);
    unsigned short* wf1_bf  = (unsigned short*)alloc((size_t)NL * 1024 * 256 * 2);
    unsigned short* wf2_bf  = (unsigned short*)alloc((size_t)NL * 256 * 1024 * 2);
    float*          xp      = (float*)alloc((size_t)MM * HH * sizeof(float));
    unsigned short* xbf     = (unsigned short*)alloc((size_t)MM * HH * 2);
    unsigned short* qkv_bf  = (unsigned short*)alloc((size_t)MM * 768 * 2);
    unsigned short* obf     = (unsigned short*)alloc((size_t)MM * HH * 2);
    unsigned short* hbf     = (unsigned short*)alloc((size_t)MM * 1024 * 2);
    float*          ybuf    = (float*)alloc((size_t)MM * HH * sizeof(float));

    {
        int n1 = NL * 768 * 256;
        cvt_bf16_kernel<<<(n1 + 255) / 256, 256, 0, stream>>>(Wqkv, wqkv_bf, n1);
        int n2 = NL * 256 * 256;
        cvt_bf16_kernel<<<(n2 + 255) / 256, 256, 0, stream>>>(Wo, wo_bf, n2);
        int n3 = NL * 1024 * 256;
        cvt_bf16_kernel<<<(n3 + 255) / 256, 256, 0, stream>>>(f_W1, wf1_bf, n3);
        int n4 = NL * 256 * 1024;
        cvt_bf16_kernel<<<(n4 + 255) / 256, 256, 0, stream>>>(f_W2, wf2_bf, n4);
    }
    scan_counts_kernel<<<1, NB, 0, stream>>>(counts, offs);
    time_kernel<<<NB, 128, 0, stream>>>(t_in, inv_freq, te_W1, te_b1,
                                        te_W2, te_b2, tp_W, tp_b, t_emb);
    {
        long long n = (long long)MM * HH;
        zero_kernel<<<(unsigned)((n + 255) / 256), 256, 0, stream>>>(xp, xbf, n);
    }
    embed_kernel<<<N, 256, 0, stream>>>(theta_t, pos_t, batch, offs,
                                        node_W, node_b, pe_W1, pe_b1,
                                        pe_W2, pe_b2, t_emb, xp, xbf);

    for (int l = 0; l < NL; ++l) {
        gemm_bf16_kernel<<<dim3(MM / 128, 768 / 64), 256, 0, stream>>>(
            xbf, wqkv_bf + (size_t)l * 768 * 256, bqkv + (size_t)l * 768,
            qkv_bf, 768, 256, OUT_BF16);
        attn_kernel<<<dim3(NHD, NB), 128, 0, stream>>>(qkv_bf, counts, obf);
        gemm_bf16_kernel<<<dim3(MM / 128, 256 / 64), 256, 0, stream>>>(
            obf, wo_bf + (size_t)l * 256 * 256, bo + (size_t)l * 256,
            ybuf, 256, 256, OUT_F32);
        resid_ln_kernel<<<MM, 256, 0, stream>>>(
            xp, ybuf, ln1_g + (size_t)l * 256, ln1_b + (size_t)l * 256, xbf);
        gemm_bf16_kernel<<<dim3(MM / 128, 1024 / 64), 256, 0, stream>>>(
            xbf, wf1_bf + (size_t)l * 1024 * 256, f_b1 + (size_t)l * 1024,
            hbf, 1024, 256, OUT_GELU);
        gemm_bf16_kernel<<<dim3(MM / 128, 256 / 64), 256, 0, stream>>>(
            hbf, wf2_bf + (size_t)l * 256 * 1024, f_b2 + (size_t)l * 256,
            ybuf, 256, 1024, OUT_F32);
        resid_ln_kernel<<<MM, 256, 0, stream>>>(
            xp, ybuf, ln2_g + (size_t)l * 256, ln2_b + (size_t)l * 256, xbf);
    }

    pool_kernel<<<NB, 256, 0, stream>>>(xp, counts, pool_W, pool_b, (float*)d_out);
}
